// TriangleAttention_9835475107907
// MI455X (gfx1250) — compile-verified, matrix operand order checked by reading, not run on hardware
//
#include <hip/hip_runtime.h>
#include <hip/hip_bf16.h>

// ---------------------------------------------------------------------------
// Causal attention (unscaled, mask-fill -1e4 == causal), FlashAttention-2
// style for gfx1250: bf16 WMMA (v_wmma_f32_16x16x32_bf16), fp32 accumulate.
// b=2, h=16, s=4096, d=128. Output (b, s, h*d) fp32.
// ---------------------------------------------------------------------------

typedef __attribute__((ext_vector_type(16))) __bf16 v16bf;
typedef __attribute__((ext_vector_type(8)))  float  v8f;

#define S_LEN   4096
#define D_DIM   128
#define H_NUM   16
#define QT      64          // query rows per workgroup (4 waves x 16)
#define KT      64          // key rows per mainloop iteration
#define NWAVE   4

// fp32 -> bf16, round-to-nearest-even
static __device__ __forceinline__ __bf16 f2bf(float f) {
  unsigned u = __builtin_bit_cast(unsigned, f);
  u += 0x7FFFu + ((u >> 16) & 1u);
  unsigned short h = (unsigned short)(u >> 16);
  return __builtin_bit_cast(__bf16, h);
}

union AFrag {
  v16bf  v;
  __bf16 e[16];
  uint4  q[2];
};

__global__ __launch_bounds__(128, 1)
void fa_causal_fwd(const float* __restrict__ Q,
                   const float* __restrict__ K,
                   const float* __restrict__ V,
                   float* __restrict__ O) {
  // Padded LDS tiles: row pitches are multiples of 16B -> aligned ds_load_b128,
  // and not multiples of 64 banks -> conflict-free fragment reads.
  __shared__ __bf16 Kl[KT][136];        // K block, [key][d]        (272B pitch)
  __shared__ __bf16 Vt[D_DIM][72];      // V block transposed [d][key] (144B pitch)
  __shared__ __bf16 Pl[NWAVE][16][72];  // per-wave P tile, [row][key]

  const int tid  = threadIdx.x;
  const int wv   = tid >> 5;            // wave id (0..3)
  const int lane = tid & 31;
  const int ln   = lane & 15;
  const int hi   = lane >> 4;           // half-wave select (K+8 / K+24 chunks)

  const int qt = blockIdx.x * QT;       // q tile base row
  const int bh = blockIdx.y;            // fused (b,h)
  const int bb = bh >> 4;
  const int hh = bh & 15;

  const float* Qb = Q + (size_t)bh * S_LEN * D_DIM;
  const float* Kb = K + (size_t)bh * S_LEN * D_DIM;
  const float* Vb = V + (size_t)bh * S_LEN * D_DIM;

  // ---- Load Q stripe once, directly into the A-matrix bf16 layout ----------
  // A 16x32 bf16 layout: lane (0..15) = row M, VGPR0..3 hold K = kb0..kb0+7,
  // VGPR4..7 hold K = kb0+16..kb0+23, where kb0 = 8*hi.
  AFrag qf[4];
  {
    const float* qr = Qb + (size_t)(qt + wv * 16 + ln) * D_DIM;
    #pragma unroll
    for (int c = 0; c < 4; ++c) {
      const int b0 = c * 32 + hi * 8;
      const int b1 = b0 + 16;
      float t0[8], t1[8];
      *(float4*)&t0[0] = *(const float4*)(qr + b0);
      *(float4*)&t0[4] = *(const float4*)(qr + b0 + 4);
      *(float4*)&t1[0] = *(const float4*)(qr + b1);
      *(float4*)&t1[4] = *(const float4*)(qr + b1 + 4);
      #pragma unroll
      for (int i = 0; i < 8; ++i) {
        qf[c].e[i]     = f2bf(t0[i]);
        qf[c].e[8 + i] = f2bf(t1[i]);
      }
    }
  }

  // Output accumulators: 16x128 fp32 per wave (8 WMMA C/D tiles).
  v8f acc[8];
  #pragma unroll
  for (int dt = 0; dt < 8; ++dt) {
    #pragma unroll
    for (int r = 0; r < 8; ++r) acc[dt][r] = 0.0f;
  }
  // Online-softmax state per C-layout row (row = r + 8*hi), replicated
  // across the 16 lanes that hold that row.
  float m_r[8], l_r[8];
  #pragma unroll
  for (int r = 0; r < 8; ++r) { m_r[r] = -INFINITY; l_r[r] = 0.0f; }

  const int nblocks = blockIdx.x + 1;   // causal: keys up to end of q tile

  for (int j = 0; j < nblocks; ++j) {
    const int kb = j * KT;
    if (j > 0) __syncthreads();         // protect Kl/Vt reuse

    // ---- Stage K (row-major) and V (transposed) into LDS as bf16 ----------
    {
      const int row  = tid >> 1;        // 0..63
      const int half = tid & 1;         // which 64-wide d half
      const float* ks = Kb + (size_t)(kb + row) * D_DIM + half * 64;
      const float* vs = Vb + (size_t)(kb + row) * D_DIM + half * 64;
      #pragma unroll
      for (int i = 0; i < 16; ++i) {
        float4 kv = *(const float4*)(ks + i * 4);
        float4 vv = *(const float4*)(vs + i * 4);
        const int d0 = half * 64 + i * 4;
        Kl[row][d0 + 0] = f2bf(kv.x);
        Kl[row][d0 + 1] = f2bf(kv.y);
        Kl[row][d0 + 2] = f2bf(kv.z);
        Kl[row][d0 + 3] = f2bf(kv.w);
        Vt[d0 + 0][row] = f2bf(vv.x);
        Vt[d0 + 1][row] = f2bf(vv.y);
        Vt[d0 + 2][row] = f2bf(vv.z);
        Vt[d0 + 3][row] = f2bf(vv.w);
      }
      if (j + 1 < nblocks) {            // prefetch next K/V block (global_prefetch_b8)
        __builtin_prefetch(Kb + (size_t)(kb + KT) * D_DIM + tid * 64, 0, 1);
        __builtin_prefetch(Vb + (size_t)(kb + KT) * D_DIM + tid * 64, 0, 1);
      }
    }
    __syncthreads();

    // ---- S = Q * K^T : 4 n-tiles x 4 k-chunks of WMMA ----------------------
    v8f sc[4];
    #pragma unroll
    for (int n = 0; n < 4; ++n) {
      #pragma unroll
      for (int r = 0; r < 8; ++r) sc[n][r] = 0.0f;
      #pragma unroll
      for (int c = 0; c < 4; ++c) {
        AFrag kf;   // B-matrix column N = key (n*16+ln): contiguous row of Kl
        kf.q[0] = *(const uint4*)&Kl[n * 16 + ln][c * 32 + hi * 8];
        kf.q[1] = *(const uint4*)&Kl[n * 16 + ln][c * 32 + 16 + hi * 8];
        sc[n] = __builtin_amdgcn_wmma_f32_16x16x32_bf16(
            false, qf[c].v, false, kf.v, (short)0, sc[n], false, false);
      }
    }

    // ---- Causal mask (diagonal blocks only) --------------------------------
    const int qw0 = qt + wv * 16;       // smallest q row in this wave
    if (kb + KT - 1 > qw0) {
      #pragma unroll
      for (int n = 0; n < 4; ++n) {
        const int kg = kb + n * 16 + ln;
        #pragma unroll
        for (int r = 0; r < 8; ++r) {
          const int qg = qw0 + r + 8 * hi;
          if (kg > qg) sc[n][r] = -10000.0f;
        }
      }
    }

    // ---- Online softmax + P tile to LDS (C-layout -> A-layout via LDS) -----
    #pragma unroll
    for (int r = 0; r < 8; ++r) {
      float mx = sc[0][r];
      #pragma unroll
      for (int n = 1; n < 4; ++n) mx = fmaxf(mx, sc[n][r]);
      #pragma unroll
      for (int off = 1; off < 16; off <<= 1)
        mx = fmaxf(mx, __shfl_xor(mx, off, 32));

      const float mnew  = fmaxf(m_r[r], mx);
      const float alpha = __expf(m_r[r] - mnew);

      float sum = 0.0f;
      #pragma unroll
      for (int n = 0; n < 4; ++n) {
        const float pv = __expf(sc[n][r] - mnew);
        sum += pv;
        Pl[wv][r + 8 * hi][n * 16 + ln] = f2bf(pv);
      }
      #pragma unroll
      for (int off = 1; off < 16; off <<= 1)
        sum += __shfl_xor(sum, off, 32);

      l_r[r] = l_r[r] * alpha + sum;
      m_r[r] = mnew;
      #pragma unroll
      for (int dt = 0; dt < 8; ++dt) acc[dt][r] *= alpha;
    }
    __syncthreads();

    // ---- O += P * V : 2 k-chunks x 8 d-tiles of WMMA -----------------------
    #pragma unroll
    for (int kc = 0; kc < 2; ++kc) {
      AFrag pa;   // A-matrix: lane row = ln, contiguous key chunks from Pl
      pa.q[0] = *(const uint4*)&Pl[wv][ln][kc * 32 + hi * 8];
      pa.q[1] = *(const uint4*)&Pl[wv][ln][kc * 32 + 16 + hi * 8];
      #pragma unroll
      for (int dt = 0; dt < 8; ++dt) {
        AFrag vf; // B-matrix column N = d (dt*16+ln): contiguous row of Vt
        vf.q[0] = *(const uint4*)&Vt[dt * 16 + ln][kc * 32 + hi * 8];
        vf.q[1] = *(const uint4*)&Vt[dt * 16 + ln][kc * 32 + 16 + hi * 8];
        acc[dt] = __builtin_amdgcn_wmma_f32_16x16x32_bf16(
            false, pa.v, false, vf.v, (short)0, acc[dt], false, false);
      }
    }
  }

  // ---- Epilogue: divide by softmax denominator, store (b, s, h*d) ----------
  #pragma unroll
  for (int r = 0; r < 8; ++r) {
    const float inv = 1.0f / l_r[r];
    const int row = qt + wv * 16 + r + 8 * hi;
    float* orow = O + ((size_t)bb * S_LEN + row) * (H_NUM * D_DIM) + hh * D_DIM;
    #pragma unroll
    for (int dt = 0; dt < 8; ++dt)
      orow[dt * 16 + ln] = acc[dt][r] * inv;
  }
}

extern "C" void kernel_launch(void* const* d_in, const int* in_sizes, int n_in,
                              void* d_out, int out_size, void* d_ws, size_t ws_size,
                              hipStream_t stream) {
  (void)in_sizes; (void)n_in; (void)out_size; (void)d_ws; (void)ws_size;
  const float* Q = (const float*)d_in[0];
  const float* K = (const float*)d_in[1];
  const float* V = (const float*)d_in[2];
  // d_in[3] (attention_mask) is pure causal: computed analytically in-kernel.
  float* O = (float*)d_out;

  dim3 grid(S_LEN / QT, 2 * H_NUM);   // 64 q-tiles x (b*h = 32)
  fa_causal_fwd<<<grid, 128, 0, stream>>>(Q, K, V, O);
}
